// Attention_56590489092170
// MI455X (gfx1250) — compile-verified, matrix operand order checked by reading, not run on hardware
//
#include <hip/hip_runtime.h>
#include <hip/hip_bf16.h>
#include <math.h>

// ---------------------------------------------------------------------------
// Causal single-head attention, B=8 S=2048 D=512, fp32 in/out. MI455X/gfx1250.
//   Kernel 0: one-shot fp32 -> bf16 conversion of x and Wq/Wk/Wv (memory bound)
//   Kernel 1: Q/K/V projection GEMMs, bf16 WMMA, 16x64 strip per wave
//   Kernel 2: flash attention; each block processes the query-tile pair
//             (mt, 127-mt) so causal (triangular) work is load-balanced.
// All matmuls on v_wmma_f32_16x16x32_bf16 (f32 accum).
// ---------------------------------------------------------------------------

typedef __attribute__((ext_vector_type(16))) __bf16          v16bf;
typedef __attribute__((ext_vector_type(16))) unsigned short  v16u;
typedef __attribute__((ext_vector_type(8)))  unsigned short  ushort8;
typedef __attribute__((ext_vector_type(8)))  float           v8f;

#define Bb      8
#define Ss      2048
#define Dd      512
#define ROWS    (Bb * Ss)                  // 16384 flattened (b,s) rows
#define INV_SQRT_D 0.044194173824159216f   // 1/sqrt(512)
#define NEG_BIG (-1.0e9f)

__device__ __forceinline__ unsigned short f2bf(float f) {
    unsigned u = __builtin_bit_cast(unsigned, f);
    u = (u + 0x7FFFu + ((u >> 16) & 1u)) >> 16;   // round-to-nearest-even
    return (unsigned short)u;
}

__device__ __forceinline__ v8f zero8() {
    v8f z;
#pragma unroll
    for (int i = 0; i < 8; ++i) z[i] = 0.0f;
    return z;
}

// A/B fragment for wmma_f32_16x16x32_bf16 from a row-major bf16 matrix with
// contiguous K. ISA layout: lane%16 = row, lane/16 = K-half:
//   elements 0..7  -> k0 + half*8 + {0..7}
//   elements 8..15 -> k0 + 16 + half*8 + {0..7}
// Two 16-byte loads, concatenated with a no-op shuffle.
__device__ __forceinline__ v16bf frag_bf16(const unsigned short* __restrict__ rowptr,
                                           int k0, int half) {
    const ushort8 s0 = *reinterpret_cast<const ushort8*>(rowptr + k0 + half * 8);
    const ushort8 s1 = *reinterpret_cast<const ushort8*>(rowptr + k0 + 16 + half * 8);
    v16u r = __builtin_shufflevector(s0, s1, 0, 1, 2, 3, 4, 5, 6, 7,
                                             8, 9, 10, 11, 12, 13, 14, 15);
    return __builtin_bit_cast(v16bf, r);
}

__device__ __forceinline__ v8f wmma_bf16(v16bf a, v16bf b, v8f c) {
    return __builtin_amdgcn_wmma_f32_16x16x32_bf16(
        /*neg_a=*/false, a, /*neg_b=*/false, b,
        /*c_mod=*/(short)0, c, /*reuse_a=*/false, /*reuse_b=*/false);
}

// ---------------------------------------------------------------------------
// Kernel 0: fp32 -> bf16 conversion, 8 elements per thread.
// ---------------------------------------------------------------------------
__global__ __launch_bounds__(256)
void cvt_bf16_kernel(const float* __restrict__ in,
                     unsigned short* __restrict__ out, int n8) {
    const int i = blockIdx.x * 256 + threadIdx.x;
    if (i >= n8) return;
    const float4 f0 = reinterpret_cast<const float4*>(in)[i * 2 + 0];
    const float4 f1 = reinterpret_cast<const float4*>(in)[i * 2 + 1];
    ushort8 o;
    o[0] = f2bf(f0.x); o[1] = f2bf(f0.y); o[2] = f2bf(f0.z); o[3] = f2bf(f0.w);
    o[4] = f2bf(f1.x); o[5] = f2bf(f1.y); o[6] = f2bf(f1.z); o[7] = f2bf(f1.w);
    reinterpret_cast<ushort8*>(out)[i] = o;
}

// ---------------------------------------------------------------------------
// Kernel 1: projections from bf16 inputs. Each wave computes a 16x64 strip
// (4 accumulator tiles, A fragment reused 4x). 8 waves per block.
// tasks per matrix = (16384/16) * (512/64) = 8192; 3 matrices -> 24576 tasks.
// mat 0 -> Q bf16 [ROWS][512]; mat 1 -> K bf16 [ROWS][512];
// mat 2 -> V^T bf16 [B][512][S].
// ---------------------------------------------------------------------------
__global__ __launch_bounds__(256)
void attn_proj_kernel(const unsigned short* __restrict__ xb,
                      const unsigned short* __restrict__ wqb,
                      const unsigned short* __restrict__ wkb,
                      const unsigned short* __restrict__ wvb,
                      const float* __restrict__ bq,
                      const float* __restrict__ bk,
                      const float* __restrict__ bv,
                      unsigned short* __restrict__ qb,
                      unsigned short* __restrict__ kb,
                      unsigned short* __restrict__ vt) {
    const int wave = threadIdx.x >> 5;
    const int lane = threadIdx.x & 31;
    const int half = lane >> 4;
    const int ln   = lane & 15;
    const int halfoff = half << 3;

    const int task = blockIdx.x * 8 + wave;            // 0 .. 24575
    const int tasks_per_mat = (ROWS / 16) * (Dd / 64); // 8192
    const int mat  = task / tasks_per_mat;
    const int rem  = task - mat * tasks_per_mat;
    const int rowt = rem >> 3;
    const int grp  = rem & 7;
    const int row0 = rowt * 16;
    const int n0g  = grp * 64;

    const unsigned short* W = (mat == 0) ? wqb : (mat == 1) ? wkb : wvb;
    const float*       bias = (mat == 0) ? bq  : (mat == 1) ? bk  : bv;

    const unsigned short* xrow = xb + (size_t)(row0 + ln) * Dd;

    v8f acc[4];
#pragma unroll
    for (int t = 0; t < 4; ++t) acc[t] = zero8();

#pragma unroll
    for (int kk = 0; kk < 16; ++kk) {
        const v16bf a = frag_bf16(xrow, kk * 32, half);
#pragma unroll
        for (int t = 0; t < 4; ++t) {
            const unsigned short* wrow = W + (size_t)(n0g + t * 16 + ln) * Dd;
            const v16bf bfr = frag_bf16(wrow, kk * 32, half);
            acc[t] = wmma_bf16(a, bfr, acc[t]);
        }
    }

#pragma unroll
    for (int t = 0; t < 4; ++t) {
        const int n = n0g + t * 16 + ln;
        const float bias_n = bias[n];
#pragma unroll
        for (int r = 0; r < 8; ++r) {
            const int gm = row0 + r + halfoff;         // flattened (b,s)
            const unsigned short h = f2bf(acc[t][r] + bias_n);
            if (mat == 0) {
                qb[(size_t)gm * Dd + n] = h;
            } else if (mat == 1) {
                kb[(size_t)gm * Dd + n] = h;
            } else {
                const int bi = gm >> 11;               // /2048
                const int s  = gm & 2047;
                vt[((size_t)bi * Dd + n) * Ss + s] = h; // V^T [B][512][S]
            }
        }
    }
}

// ---------------------------------------------------------------------------
// Kernel 2: flash attention for one 16-row query tile. 128 threads / 4 waves.
// Wave w: score tile at key cols j*64 + w*16; O slice cols [w*128, w*128+128).
// Softmax: all 128 threads, 8 elems each, 8-lane shfl_xor row reductions.
// ---------------------------------------------------------------------------
__device__ __forceinline__
void attn_tile_body(int m0,
                    const unsigned short* __restrict__ Qbatch,
                    const unsigned short* __restrict__ Kbase,
                    const unsigned short* __restrict__ Vbase,
                    const unsigned char*  __restrict__ padb,
                    float* __restrict__ obatch,
                    unsigned short* qs, float (*s_tile)[64],
                    unsigned short* p_tile,
                    float* row_m, float* row_l, float* row_c, float* row_linv) {
    const int tid  = threadIdx.x;
    const int wave = tid >> 5;
    const int lane = tid & 31;
    const int half = lane >> 4;
    const int halfoff = half << 3;
    const int ln   = lane & 15;

    const unsigned short* Qbase = Qbatch + (size_t)m0 * Dd;

    // stage the 16 contiguous Q rows into LDS (8192 ushorts)
    for (int i = tid * 8; i < 16 * Dd; i += 128 * 8)
        *reinterpret_cast<ushort8*>(qs + i) =
            *reinterpret_cast<const ushort8*>(Qbase + i);
    if (tid < 16) { row_m[tid] = -1.0e30f; row_l[tid] = 0.0f; }
    __syncthreads();

    v8f acc[8];
#pragma unroll
    for (int t = 0; t < 8; ++t) acc[t] = zero8();
    const int ncol0 = wave * 128;

    // softmax work distribution: row = tid/8, 8-col segment = tid%8
    const int srow = tid >> 3;
    const int sseg = tid & 7;

    const int jmax = (m0 + 15) >> 6;       // causal bound on 64-wide key blocks
    for (int j = 0; j <= jmax; ++j) {
        // ---- score phase: S_tile[:, wave*16 .. +15] over full d=512
        v8f sacc = zero8();
        const int krow = j * 64 + wave * 16 + ln;
        const unsigned short* krp = Kbase + (size_t)krow * Dd;
#pragma unroll
        for (int kk = 0; kk < 16; ++kk) {
            const v16bf a  = frag_bf16(qs + ln * Dd, kk * 32, half);
            const v16bf bf = frag_bf16(krp, kk * 32, half);
            sacc = wmma_bf16(a, bf, sacc);
        }
        if (j < jmax)   // pull next key block toward the caches
            __builtin_prefetch(Kbase + (size_t)(krow + 64) * Dd, 0, 1);

        const int coln = j * 64 + wave * 16 + ln;
        const bool padc = (padb[coln] != 0);
#pragma unroll
        for (int r = 0; r < 8; ++r) {
            const int gm = m0 + r + halfoff;
            float sv = sacc[r] * INV_SQRT_D;
            if (coln > gm || padc) sv = NEG_BIG;
            s_tile[r + halfoff][wave * 16 + ln] = sv;
        }
        __syncthreads();

        // ---- online softmax: 128 threads, 8 elems each
        {
            float sv[8];
#pragma unroll
            for (int c = 0; c < 8; ++c) sv[c] = s_tile[srow][sseg * 8 + c];
            float mx = sv[0];
#pragma unroll
            for (int c = 1; c < 8; ++c) mx = fmaxf(mx, sv[c]);
            // reduce max over the 8-lane row group
            mx = fmaxf(mx, __shfl_xor(mx, 1));
            mx = fmaxf(mx, __shfl_xor(mx, 2));
            mx = fmaxf(mx, __shfl_xor(mx, 4));
            const float m_old = row_m[srow];
            mx = fmaxf(mx, m_old);
            const float corr = __expf(m_old - mx);

            float sum = 0.0f;
            ushort8 pv;
#pragma unroll
            for (int c = 0; c < 8; ++c) {
                const float p = __expf(sv[c] - mx);
                sum += p;
                pv[c] = f2bf(p);
            }
            *reinterpret_cast<ushort8*>(p_tile + srow * 64 + sseg * 8) = pv;
            // reduce sum over the 8-lane row group
            sum += __shfl_xor(sum, 1);
            sum += __shfl_xor(sum, 2);
            sum += __shfl_xor(sum, 4);
            if (sseg == 0) {
                row_l[srow] = row_l[srow] * corr + sum;
                row_m[srow] = mx;
                row_c[srow] = corr;
            }
        }
        __syncthreads();

        // ---- PV phase: rescale accumulators, then O += P @ V
        float cr[8];
#pragma unroll
        for (int r = 0; r < 8; ++r) cr[r] = row_c[r + halfoff];
#pragma unroll
        for (int t = 0; t < 8; ++t)
#pragma unroll
            for (int r = 0; r < 8; ++r) acc[t][r] *= cr[r];

#pragma unroll
        for (int kk2 = 0; kk2 < 2; ++kk2) {
            const v16bf a = frag_bf16(p_tile + ln * 64, kk2 * 32, half);
#pragma unroll
            for (int t = 0; t < 8; ++t) {
                const unsigned short* vrp =
                    Vbase + (size_t)(ncol0 + t * 16 + ln) * Ss;
                const v16bf bf = frag_bf16(vrp, j * 64 + kk2 * 32, half);
                acc[t] = wmma_bf16(a, bf, acc[t]);
            }
        }
        __syncthreads();   // protect s_tile / p_tile for next iteration
    }

    if (tid < 16) row_linv[tid] = 1.0f / row_l[tid];
    __syncthreads();

    float li[8];
#pragma unroll
    for (int r = 0; r < 8; ++r) li[r] = row_linv[r + halfoff];

    float* ob = obatch + (size_t)m0 * Dd;
#pragma unroll
    for (int t = 0; t < 8; ++t)
#pragma unroll
        for (int r = 0; r < 8; ++r)
            ob[(size_t)(r + halfoff) * Dd + ncol0 + t * 16 + ln] =
                acc[t][r] * li[r];
}

// Block = (b, mt-pair). Processes query tiles mt and 127-mt so the causal
// triangle is load-balanced: every block runs 33 key-block iterations.
__global__ __launch_bounds__(128)
void attn_fa_kernel(const unsigned short* __restrict__ qb,
                    const unsigned short* __restrict__ kb,
                    const unsigned short* __restrict__ vt,
                    const unsigned char*  __restrict__ pad,
                    float* __restrict__ out) {
    __shared__ __align__(16) unsigned short qs[16 * Dd];      // 16 KB
    __shared__ float           s_tile[16][64];                // 4 KB
    __shared__ __align__(16) unsigned short p_tile[16 * 64];  // 2 KB
    __shared__ float           row_m[16];
    __shared__ float           row_l[16];
    __shared__ float           row_c[16];
    __shared__ float           row_linv[16];

    const int b   = blockIdx.x >> 6;       // 64 tile-pairs per batch
    const int mtp = blockIdx.x & 63;

    const unsigned short* Qbatch = qb + (size_t)b * Ss * Dd;
    const unsigned short* Kbase  = kb + (size_t)b * Ss * Dd;
    const unsigned short* Vbase  = vt + (size_t)b * Dd * Ss;
    const unsigned char*  padb   = pad + (size_t)b * Ss;
    float*                obatch = out + (size_t)b * Ss * Dd;

    attn_tile_body(mtp * 16, Qbatch, Kbase, Vbase, padb, obatch,
                   qs, s_tile, p_tile, row_m, row_l, row_c, row_linv);
    attn_tile_body((127 - mtp) * 16, Qbatch, Kbase, Vbase, padb, obatch,
                   qs, s_tile, p_tile, row_m, row_l, row_c, row_linv);
}

// ---------------------------------------------------------------------------
// Launch. Inputs: x, pad_mask(bool/u8), Wq, bq, Wk, bk, Wv, bv.
// ws layout (ushort units):
//   xb[8388608] | wq[262144] | wk[262144] | wv[262144] |
//   qb[8388608] | kb[8388608] | vt[8388608]            (~69 MB total)
// ---------------------------------------------------------------------------
extern "C" void kernel_launch(void* const* d_in, const int* in_sizes, int n_in,
                              void* d_out, int out_size, void* d_ws, size_t ws_size,
                              hipStream_t stream) {
    (void)in_sizes; (void)n_in; (void)out_size; (void)ws_size;

    const float*         x   = (const float*)d_in[0];
    const unsigned char* pad = (const unsigned char*)d_in[1];
    const float*         Wq  = (const float*)d_in[2];
    const float*         bq  = (const float*)d_in[3];
    const float*         Wk  = (const float*)d_in[4];
    const float*         bk  = (const float*)d_in[5];
    const float*         Wv  = (const float*)d_in[6];
    const float*         bv  = (const float*)d_in[7];
    float*               out = (float*)d_out;

    const size_t xElems = (size_t)ROWS * Dd;   // 8,388,608
    const size_t wElems = (size_t)Dd * Dd;     //   262,144
    unsigned short* xb  = (unsigned short*)d_ws;
    unsigned short* wqb = xb  + xElems;
    unsigned short* wkb = wqb + wElems;
    unsigned short* wvb = wkb + wElems;
    unsigned short* qb  = wvb + wElems;
    unsigned short* kb  = qb  + xElems;
    unsigned short* vt  = kb  + xElems;

    // Kernel 0: one-shot bf16 conversion (8 elems/thread)
    cvt_bf16_kernel<<<(int)(xElems / 8 / 256), 256, 0, stream>>>(x,  xb,  (int)(xElems / 8));
    cvt_bf16_kernel<<<(int)(wElems / 8 / 256), 256, 0, stream>>>(Wq, wqb, (int)(wElems / 8));
    cvt_bf16_kernel<<<(int)(wElems / 8 / 256), 256, 0, stream>>>(Wk, wkb, (int)(wElems / 8));
    cvt_bf16_kernel<<<(int)(wElems / 8 / 256), 256, 0, stream>>>(Wv, wvb, (int)(wElems / 8));

    // Kernel 1: 24576 wave-tasks / 8 waves per block = 3072 blocks
    attn_proj_kernel<<<3072, 256, 0, stream>>>(xb, wqb, wkb, wvb, bq, bk, bv,
                                               qb, kb, vt);
    // Kernel 2: B * 64 tile-pairs = 512 blocks of 128 threads
    attn_fa_kernel<<<Bb * 64, 128, 0, stream>>>(qb, kb, vt, pad, out);
}
